// AriaExperts_6871947674156
// MI455X (gfx1250) — compile-verified, matrix operand order checked by reading, not run on hardware
//
#include <hip/hip_runtime.h>
#include <hip/hip_bf16.h>

#define NUM_TOKENS 2048
#define HIDDEN     1024
#define INTER      2048
#define NUM_EXPERTS 8
#define TOPK        2
#define FLAT_ROWS  (NUM_TOKENS * TOPK)   // 4096
#define W1ROW      (2 * INTER)           // 4096

#define MT 128          // rows per block tile
#define NT 128          // cols per block tile
#define KT 32           // K per WMMA step (f16)
#define LDS_STRIDE 40   // halves per LDS row (80B: 16B frag alignment, bank spread)
#define MAX_TILES (FLAT_ROWS / MT)       // 32 row tiles max per expert

typedef __attribute__((ext_vector_type(16))) _Float16 v16h;
typedef __attribute__((ext_vector_type(8)))  _Float16 v8h;
typedef __attribute__((ext_vector_type(4)))  _Float16 v4h;
typedef __attribute__((ext_vector_type(2)))  _Float16 v2h;
typedef __attribute__((ext_vector_type(8)))  float    v8f;

// ---------------------------------------------------------------- init
__global__ void moe_init(float* __restrict__ out, int* __restrict__ counters) {
    int i = blockIdx.x * blockDim.x + threadIdx.x;
    if (i < NUM_TOKENS * HIDDEN) out[i] = 0.f;
    if (i < 16) counters[i] = 0;          // counts[8] + cursors[8], contiguous
}

// ---------------------------------------------------------------- routing
__global__ void moe_router(const float* __restrict__ logits,
                           float* __restrict__ scoreFlat,
                           int* __restrict__ idxFlat,
                           int* __restrict__ counts) {
    int t = blockIdx.x * blockDim.x + threadIdx.x;
    if (t >= NUM_TOKENS) return;
    float l[NUM_EXPERTS];
#pragma unroll
    for (int e = 0; e < NUM_EXPERTS; ++e) l[e] = logits[t * NUM_EXPERTS + e];
    int i0 = 0; float b0 = l[0];
#pragma unroll
    for (int e = 1; e < NUM_EXPERTS; ++e) if (l[e] > b0) { b0 = l[e]; i0 = e; }
    int i1 = -1; float b1 = -3.4e38f;
#pragma unroll
    for (int e = 0; e < NUM_EXPERTS; ++e)
        if (e != i0 && l[e] > b1) { b1 = l[e]; i1 = e; }
    float r = expf(b1 - b0);               // softmax over the 2 picks (b0 >= b1)
    float inv = 1.f / (1.f + r);
    idxFlat[2 * t + 0] = i0;  scoreFlat[2 * t + 0] = inv;
    idxFlat[2 * t + 1] = i1;  scoreFlat[2 * t + 1] = r * inv;
    atomicAdd(&counts[i0], 1);
    atomicAdd(&counts[i1], 1);
}

__global__ void moe_scan(const int* __restrict__ counts,
                         int* __restrict__ offsets,
                         int* __restrict__ cursors) {
    if (threadIdx.x == 0 && blockIdx.x == 0) {
        int run = 0;
        for (int e = 0; e < NUM_EXPERTS; ++e) {
            offsets[e] = run;
            cursors[e] = run;
            run += counts[e];
        }
        offsets[NUM_EXPERTS] = run;        // == FLAT_ROWS
    }
}

__global__ void moe_scatter(const int* __restrict__ idxFlat,
                            int* __restrict__ cursors,
                            int* __restrict__ rowsBuf) {
    int f = blockIdx.x * blockDim.x + threadIdx.x;
    if (f >= FLAT_ROWS) return;
    int e = idxFlat[f];
    int pos = atomicAdd(&cursors[e], 1);
    rowsBuf[pos] = f;                      // f = token*2 + k slot
}

// ---------------------------------------------------------------- fc1 + SwiGLU
__global__ __launch_bounds__(256) void moe_fc1(
    const float* __restrict__ X, const float* __restrict__ W1,
    const int* __restrict__ rowsBuf, const int* __restrict__ offsets,
    _Float16* __restrict__ ACT) {

    const int e    = blockIdx.y >> 5;          // / MAX_TILES
    const int tile = blockIdx.y & (MAX_TILES - 1);
    const int start = offsets[e] + tile * MT;
    const int end   = offsets[e + 1];
    if (start >= end) return;
    int mEnd = end - start; if (mEnd > MT) mEnd = MT;
    const int n0 = blockIdx.x * NT;            // column tile within INTER

    __shared__ _Float16 As[2][MT][LDS_STRIDE];
    __shared__ _Float16 Bp[2][NT][LDS_STRIDE];
    __shared__ _Float16 Bg[2][NT][LDS_STRIDE];
    __shared__ int sRow[MT];

    const int tid = threadIdx.x;
    if (tid < MT) sRow[tid] = (tid < mEnd) ? rowsBuf[start + tid] : -1;
    __syncthreads();

    const float* W1e = W1 + (size_t)e * HIDDEN * W1ROW;

    // --- per-thread staging geometry (hoisted; no branches in the K loop) ---
    const int mSubA = tid >> 3;                // A: 0..31 (+p*32)
    const int kA    = (tid & 7) * 4;           // A: 0..28
    const float* aSrc[4];
#pragma unroll
    for (int p = 0; p < 4; ++p) {
        const int f = sRow[p * 32 + mSubA];
        const int token = (f >= 0) ? (f >> 1) : 0;   // clamp: garbage rows never stored
        aSrc[p] = X + (size_t)token * HIDDEN + kA;
    }
    const int krB = tid >> 5;                  // B: k-pair id 0..7 (+p*8)
    const int ncB = (tid & 31) * 4;            // B: 0..124
    const float* bSrc[2];
#pragma unroll
    for (int p = 0; p < 2; ++p)
        bSrc[p] = W1e + (size_t)(2 * (p * 8 + krB)) * W1ROW + n0 + ncB;

    const int lane = tid & 31;
    const int wave = tid >> 5;
    const int wm = (wave & 1) * 64;
    const int wn = (wave >> 1) * 32;
    const int l16 = lane & 15;
    const int kb  = (lane >> 4) * 8;

    v8f accP[4][2], accG[4][2];
#pragma unroll
    for (int im = 0; im < 4; ++im)
#pragma unroll
        for (int jn = 0; jn < 2; ++jn)
#pragma unroll
            for (int r = 0; r < 8; ++r) { accP[im][jn][r] = 0.f; accG[im][jn][r] = 0.f; }

    float4 aReg[4], bpR[2][2], bgR[2][2];

    auto loadStage = [&](int k0) {
#pragma unroll
        for (int p = 0; p < 4; ++p)
            aReg[p] = *(const float4*)(aSrc[p] + k0);
#pragma unroll
        for (int p = 0; p < 2; ++p) {
            const float* s = bSrc[p] + (size_t)k0 * W1ROW;
            bpR[p][0] = *(const float4*)s;
            bpR[p][1] = *(const float4*)(s + W1ROW);
            bgR[p][0] = *(const float4*)(s + INTER);
            bgR[p][1] = *(const float4*)(s + INTER + W1ROW);
        }
    };
    auto storeStage = [&](int buf) {
#pragma unroll
        for (int p = 0; p < 4; ++p) {
            v4h h; h[0] = (_Float16)aReg[p].x; h[1] = (_Float16)aReg[p].y;
                   h[2] = (_Float16)aReg[p].z; h[3] = (_Float16)aReg[p].w;
            *(v4h*)&As[buf][p * 32 + mSubA][kA] = h;
        }
#pragma unroll
        for (int p = 0; p < 2; ++p) {
            const int kp = 2 * (p * 8 + krB);
            const float* p0 = (const float*)&bpR[p][0];
            const float* p1 = (const float*)&bpR[p][1];
            const float* g0 = (const float*)&bgR[p][0];
            const float* g1 = (const float*)&bgR[p][1];
#pragma unroll
            for (int j = 0; j < 4; ++j) {
                v2h hp; hp[0] = (_Float16)p0[j]; hp[1] = (_Float16)p1[j];
                *(v2h*)&Bp[buf][ncB + j][kp] = hp;
                v2h hg; hg[0] = (_Float16)g0[j]; hg[1] = (_Float16)g1[j];
                *(v2h*)&Bg[buf][ncB + j][kp] = hg;
            }
        }
    };
    auto compute = [&](int buf) {
        v16h a[4];
#pragma unroll
        for (int im = 0; im < 4; ++im) {
            const _Float16* p = &As[buf][wm + im * 16 + l16][kb];
            v8h lo = *(const v8h*)p;
            v8h hi = *(const v8h*)(p + 16);
            a[im] = __builtin_shufflevector(lo, hi, 0,1,2,3,4,5,6,7,8,9,10,11,12,13,14,15);
        }
#pragma unroll
        for (int jn = 0; jn < 2; ++jn) {
            const _Float16* pb = &Bp[buf][wn + jn * 16 + l16][kb];
            v8h plo = *(const v8h*)pb, phi = *(const v8h*)(pb + 16);
            v16h bp = __builtin_shufflevector(plo, phi, 0,1,2,3,4,5,6,7,8,9,10,11,12,13,14,15);
            const _Float16* pg = &Bg[buf][wn + jn * 16 + l16][kb];
            v8h glo = *(const v8h*)pg, ghi = *(const v8h*)(pg + 16);
            v16h bg = __builtin_shufflevector(glo, ghi, 0,1,2,3,4,5,6,7,8,9,10,11,12,13,14,15);
#pragma unroll
            for (int im = 0; im < 4; ++im) {
                accP[im][jn] = __builtin_amdgcn_wmma_f32_16x16x32_f16(
                    false, a[im], false, bp, (short)0, accP[im][jn], false, false);
                accG[im][jn] = __builtin_amdgcn_wmma_f32_16x16x32_f16(
                    false, a[im], false, bg, (short)0, accG[im][jn], false, false);
            }
        }
    };

    // --- software pipeline: load(k+1) || compute(k), double-buffered LDS ---
    loadStage(0);
    storeStage(0);
    int cur = 0;
    for (int k0 = 0; k0 < HIDDEN; k0 += KT) {
        const bool more = (k0 + KT) < HIDDEN;
        if (more) loadStage(k0 + KT);      // global loads in flight during compute
        __syncthreads();                   // buffer `cur` fully staged
        compute(cur);
        if (more) storeStage(cur ^ 1);     // write other buffer: no barrier needed
        cur ^= 1;
    }

    // ---- epilogue: SwiGLU, store f16 activations ----
    const int mh = (lane >> 4) * 8;
#pragma unroll
    for (int im = 0; im < 4; ++im)
#pragma unroll
        for (int jn = 0; jn < 2; ++jn)
#pragma unroll
            for (int r = 0; r < 8; ++r) {
                const int m = wm + im * 16 + mh + r;
                if (m < mEnd) {
                    const float pv = accP[im][jn][r];
                    const float gv = accG[im][jn][r];
                    const float s  = pv / (1.f + __expf(-pv));   // silu
                    ACT[(size_t)(start + m) * INTER + n0 + wn + jn * 16 + l16] =
                        (_Float16)(s * gv);
                }
            }
}

// ---------------------------------------------------------------- fc2 + combine
__global__ __launch_bounds__(256) void moe_fc2(
    const _Float16* __restrict__ ACT, const float* __restrict__ W2,
    const int* __restrict__ rowsBuf, const int* __restrict__ offsets,
    const float* __restrict__ scoreFlat, float* __restrict__ OUT) {

    const int e    = blockIdx.y >> 5;
    const int tile = blockIdx.y & (MAX_TILES - 1);
    const int start = offsets[e] + tile * MT;
    const int end   = offsets[e + 1];
    if (start >= end) return;
    int mEnd = end - start; if (mEnd > MT) mEnd = MT;
    const int n0 = blockIdx.x * NT;            // column tile within HIDDEN

    __shared__ _Float16 As[2][MT][LDS_STRIDE];
    __shared__ _Float16 Bs[2][NT][LDS_STRIDE];
    __shared__ int   sRow[MT];
    __shared__ float sScore[MT];

    const int tid = threadIdx.x;
    if (tid < MT) {
        int f = (tid < mEnd) ? rowsBuf[start + tid] : -1;
        sRow[tid]   = f;
        sScore[tid] = (f >= 0) ? scoreFlat[f] : 0.f;
    }
    __syncthreads();

    const float* W2e = W2 + (size_t)e * INTER * HIDDEN;

    // --- per-thread staging geometry (hoisted, branch-free) ---
    const int mSubA = tid >> 2;                // A: 0..63 (+p*64)
    const int kA    = (tid & 3) * 8;           // A: 0..24
    const _Float16* aSrc[2];
#pragma unroll
    for (int p = 0; p < 2; ++p) {
        int row = start + p * 64 + mSubA;      // clamp: rows past end never stored
        if (row > FLAT_ROWS - 1) row = FLAT_ROWS - 1;
        aSrc[p] = ACT + (size_t)row * INTER + kA;
    }
    const int krB = tid >> 5;                  // k-pair id 0..7 (+p*8)
    const int ncB = (tid & 31) * 4;
    const float* bSrc[2];
#pragma unroll
    for (int p = 0; p < 2; ++p)
        bSrc[p] = W2e + (size_t)(2 * (p * 8 + krB)) * HIDDEN + n0 + ncB;

    const int lane = tid & 31;
    const int wave = tid >> 5;
    const int wm = (wave & 1) * 64;
    const int wn = (wave >> 1) * 32;
    const int l16 = lane & 15;
    const int kb  = (lane >> 4) * 8;

    v8f acc[4][2];
#pragma unroll
    for (int im = 0; im < 4; ++im)
#pragma unroll
        for (int jn = 0; jn < 2; ++jn)
#pragma unroll
            for (int r = 0; r < 8; ++r) acc[im][jn][r] = 0.f;

    v8h aReg[2];
    float4 bR[2][2];

    auto loadStage = [&](int k0) {
#pragma unroll
        for (int p = 0; p < 2; ++p)
            aReg[p] = *(const v8h*)(aSrc[p] + k0);
#pragma unroll
        for (int p = 0; p < 2; ++p) {
            const float* s = bSrc[p] + (size_t)k0 * HIDDEN;
            bR[p][0] = *(const float4*)s;
            bR[p][1] = *(const float4*)(s + HIDDEN);
        }
    };
    auto storeStage = [&](int buf) {
#pragma unroll
        for (int p = 0; p < 2; ++p)
            *(v8h*)&As[buf][p * 64 + mSubA][kA] = aReg[p];
#pragma unroll
        for (int p = 0; p < 2; ++p) {
            const int kp = 2 * (p * 8 + krB);
            const float* b0 = (const float*)&bR[p][0];
            const float* b1 = (const float*)&bR[p][1];
#pragma unroll
            for (int j = 0; j < 4; ++j) {
                v2h h; h[0] = (_Float16)b0[j]; h[1] = (_Float16)b1[j];
                *(v2h*)&Bs[buf][ncB + j][kp] = h;
            }
        }
    };
    auto compute = [&](int buf) {
        v16h a[4];
#pragma unroll
        for (int im = 0; im < 4; ++im) {
            const _Float16* p = &As[buf][wm + im * 16 + l16][kb];
            v8h lo = *(const v8h*)p;
            v8h hi = *(const v8h*)(p + 16);
            a[im] = __builtin_shufflevector(lo, hi, 0,1,2,3,4,5,6,7,8,9,10,11,12,13,14,15);
        }
#pragma unroll
        for (int jn = 0; jn < 2; ++jn) {
            const _Float16* pb = &Bs[buf][wn + jn * 16 + l16][kb];
            v8h blo = *(const v8h*)pb, bhi = *(const v8h*)(pb + 16);
            v16h b = __builtin_shufflevector(blo, bhi, 0,1,2,3,4,5,6,7,8,9,10,11,12,13,14,15);
#pragma unroll
            for (int im = 0; im < 4; ++im)
                acc[im][jn] = __builtin_amdgcn_wmma_f32_16x16x32_f16(
                    false, a[im], false, b, (short)0, acc[im][jn], false, false);
        }
    };

    loadStage(0);
    storeStage(0);
    int cur = 0;
    for (int k0 = 0; k0 < INTER; k0 += KT) {
        const bool more = (k0 + KT) < INTER;
        if (more) loadStage(k0 + KT);
        __syncthreads();
        compute(cur);
        if (more) storeStage(cur ^ 1);
        cur ^= 1;
    }

    // ---- epilogue: score-weighted combine (TOPK=2 commutative adds/element) ----
    const int mh = (lane >> 4) * 8;
#pragma unroll
    for (int im = 0; im < 4; ++im)
#pragma unroll
        for (int jn = 0; jn < 2; ++jn)
#pragma unroll
            for (int r = 0; r < 8; ++r) {
                const int m = wm + im * 16 + mh + r;
                if (m < mEnd) {
                    const int token = sRow[m] >> 1;
                    atomicAdd(&OUT[(size_t)token * HIDDEN + n0 + wn + jn * 16 + l16],
                              sScore[m] * acc[im][jn][r]);
                }
            }
}

// ---------------------------------------------------------------- launch
extern "C" void kernel_launch(void* const* d_in, const int* in_sizes, int n_in,
                              void* d_out, int out_size, void* d_ws, size_t ws_size,
                              hipStream_t stream) {
    const float* X  = (const float*)d_in[0];   // [2048, 1024]
    const float* RL = (const float*)d_in[1];   // [2048, 8]
    const float* W1 = (const float*)d_in[2];   // [8, 1024, 4096]
    const float* W2 = (const float*)d_in[3];   // [8, 2048, 1024]
    float* OUT = (float*)d_out;                // [2048, 1024]

    char* w = (char*)d_ws;
    _Float16* ACT    = (_Float16*)w;  w += (size_t)FLAT_ROWS * INTER * sizeof(_Float16); // 16 MB
    int*   rowsBuf   = (int*)w;       w += FLAT_ROWS * sizeof(int);
    float* scoreFlat = (float*)w;     w += FLAT_ROWS * sizeof(float);
    int*   idxFlat   = (int*)w;       w += FLAT_ROWS * sizeof(int);
    int*   counts    = (int*)w;       w += NUM_EXPERTS * sizeof(int);
    int*   cursors   = (int*)w;       w += NUM_EXPERTS * sizeof(int);
    int*   offsets   = (int*)w;       w += (NUM_EXPERTS + 1) * sizeof(int);
    (void)in_sizes; (void)n_in; (void)out_size; (void)ws_size;

    moe_init   <<<(NUM_TOKENS * HIDDEN + 255) / 256, 256, 0, stream>>>(OUT, counts);
    moe_router <<<(NUM_TOKENS + 255) / 256, 256, 0, stream>>>(RL, scoreFlat, idxFlat, counts);
    moe_scan   <<<1, 1, 0, stream>>>(counts, offsets, cursors);
    moe_scatter<<<(FLAT_ROWS + 255) / 256, 256, 0, stream>>>(idxFlat, cursors, rowsBuf);

    moe_fc1<<<dim3(INTER / NT, NUM_EXPERTS * MAX_TILES), 256, 0, stream>>>(
        X, W1, rowsBuf, offsets, ACT);
    moe_fc2<<<dim3(HIDDEN / NT, NUM_EXPERTS * MAX_TILES), 256, 0, stream>>>(
        ACT, W2, rowsBuf, offsets, scoreFlat, OUT);
}
